// LSTM_predictor_cell_34162169873137
// MI455X (gfx1250) — matrix operand order repaired and checked by reading, not compiled, not run on hardware
//
#include <hip/hip_runtime.h>
#include <cstdint>
#include <cstddef>

// ---------------------------------------------------------------------------
// Types for CDNA5 WMMA (wave32)
// ---------------------------------------------------------------------------
typedef __attribute__((ext_vector_type(16))) __bf16 v16bf;
typedef __attribute__((ext_vector_type(8)))  __bf16 v8bf;
typedef __attribute__((ext_vector_type(8)))  float  v8f;
typedef __attribute__((ext_vector_type(8)))  float  f32x8;
typedef __attribute__((ext_vector_type(4)))  unsigned int u32x4;
typedef __attribute__((ext_vector_type(8)))  int    i32x8;
typedef __attribute__((ext_vector_type(4)))  int    i32x4;

#define B_  256
#define T_  512
#define F_  64
#define H_  1024
#define L4_ 4096
#define FUT_ 6
#define TO_ (T_ + FUT_)

#if __has_builtin(__builtin_amdgcn_tensor_load_to_lds) && __has_builtin(__builtin_amdgcn_s_wait_tensorcnt)
#define USE_TDM 1
#else
#define USE_TDM 0
#endif

// ---------------------------------------------------------------------------
// TDM: 2-D tile (tile_k elements contiguous x tile_rows rows) global -> LDS.
// D# layout per CDNA5 ISA 8.3/8.4 (group0: count/lds/global/type,
// group1: data_size=2B, dims, tile dims, dim0 stride). clang-23 exposes the
// 6-arg builtin; the extra i32x8 is the upper-dim group, zero for 2-D tiles.
// ---------------------------------------------------------------------------
#if USE_TDM
static __device__ __forceinline__ void tdm_load2d_bf16(
    uint32_t lds_byte_off, const void* gptr,
    uint32_t tile_k, uint32_t tile_rows,
    uint32_t row_stride_elems, uint32_t tensor_d0, uint32_t tensor_d1)
{
  uint64_t ga = (uint64_t)(uintptr_t)gptr;
  u32x4 g0 = { 1u,                                  // count=1, user descriptor
               lds_byte_off,
               (uint32_t)(ga & 0xffffffffu),        // global_addr[31:0]
               (uint32_t)((ga >> 32) & 0x1ffffffu) | (2u << 30) }; // addr[56:32] | type=2
  i32x8 g1 = { (int)(1u << 16),                                       // data_size=1 -> 2 bytes
               (int)((tensor_d0 & 0xffffu) << 16),                    // tensor_dim0[15:0]
               (int)(((tensor_d0 >> 16) & 0xffffu) |
                     ((tensor_d1 & 0xffffu) << 16)),                  // dim0 hi | dim1 lo
               (int)(((tensor_d1 >> 16) & 0xffffu) |
                     ((tile_k & 0xffffu) << 16)),                     // dim1 hi | tile_dim0
               (int)(tile_rows & 0xffffu),                            // tile_dim1 (tile_dim2=0)
               (int)row_stride_elems,                                 // tensor_dim0_stride[31:0]
               0, 0 };
  i32x4 g2 = { 0, 0, 0, 0 };                        // dims 2/3 unused (2-D tile)
  i32x4 g3 = { 0, 0, 0, 0 };                        // dim 4 unused
  i32x8 g4 = { 0, 0, 0, 0, 0, 0, 0, 0 };            // upper-dim group, unused
  __builtin_amdgcn_tensor_load_to_lds(g0, g1, g2, g3, g4, 0);
}
#endif

// ---------------------------------------------------------------------------
// Prep kernels
// ---------------------------------------------------------------------------
__global__ void k_bias_combine(const float* bi0, const float* bh0,
                               const float* bi1, const float* bh1,
                               float* b0, float* b1)
{
  int i = blockIdx.x * blockDim.x + threadIdx.x;
  if (i < L4_) { b0[i] = bi0[i] + bh0[i]; b1[i] = bi1[i] + bh1[i]; }
}

// Concatenate [s0 (N x K0) | s1 (N x K1)] row-wise into bf16 dst (N x (K0+K1))
__global__ void k_wcat_bf16(const float* s0, int K0, const float* s1, int K1,
                            __bf16* dst, int N)
{
  const int K = K0 + K1;
  const long total = (long)N * K;
  for (long idx = (long)blockIdx.x * blockDim.x + threadIdx.x; idx < total;
       idx += (long)gridDim.x * blockDim.x) {
    long n = idx / K, k = idx % K;
    float v = (k < K0) ? s0[n * K0 + k] : s1[n * K1 + (k - K0)];
    dst[idx] = (__bf16)v;
  }
}

__global__ void k_zero_f32(float* p, long n)
{
  for (long i = (long)blockIdx.x * blockDim.x + threadIdx.x; i < n;
       i += (long)gridDim.x * blockDim.x) p[i] = 0.0f;
}

// ---------------------------------------------------------------------------
// Gate GEMM:  G(256 x 4096) = [A0 (256xK0) | A1 (256xK1)] @ W^T + bias
// W is bf16, row-major (4096 x (K0+K1)). Block = 8 waves, tile 128(M) x 64(N).
// The 64x32 B tile is staged into LDS by the TDM, double-buffered so the DMA
// for chunk i+1 overlaps the WMMAs of chunk i (one barrier per chunk).
// ---------------------------------------------------------------------------
__global__ __launch_bounds__(256) void k_lstm_gates(
    const float* __restrict__ A0, int lda0, int K0,
    const float* __restrict__ A1, int lda1, int K1,
    const __bf16* __restrict__ W, const float* __restrict__ bias,
    float* __restrict__ G)
{
  __shared__ __bf16 tileB[2 * 64 * 32];   // 2 x 4 KB double buffer

  const int lane = threadIdx.x & 31;
  const int wave = threadIdx.x >> 5;
  const int n0   = blockIdx.x * 64;
  const int m0   = blockIdx.y * 128 + wave * 16;
  const int Ktot = K0 + K1;
  const int nch  = Ktot >> 5;

  const int am   = m0 + (lane & 15);        // A fragment row
  const int akb  = (lane >> 4) * 16;        // A K sub-offset (contiguous, matches B)
  const int bk16 = (lane >> 4) * 16;        // B K sub-offset (contiguous, matches A)

  v8f acc[4] = {};

#if USE_TDM
  const uint32_t ldso = (uint32_t)(uintptr_t)(&tileB[0]);
  // prologue: fill buffer 0 with chunk 0
  if (threadIdx.x == 0) {
    tdm_load2d_bf16(ldso, W + (size_t)n0 * Ktot, 32, 64,
                    (uint32_t)Ktot, (uint32_t)Ktot, L4_);
  }
  __builtin_amdgcn_s_wait_tensorcnt(0);
  __syncthreads();
#endif

  for (int i = 0; i < nch; ++i) {
    const int k0 = i << 5;

#if USE_TDM
    // kick off DMA for the next chunk into the other buffer (overlaps compute)
    if (threadIdx.x == 0 && (i + 1) < nch) {
      tdm_load2d_bf16(ldso + ((unsigned)(i + 1) & 1u) * 4096u,
                      W + (size_t)n0 * Ktot + (k0 + 32), 32, 64,
                      (uint32_t)Ktot, (uint32_t)Ktot, L4_);
    }
    const __bf16* tb = &tileB[(i & 1) * (64 * 32)];
#else
    for (int e = threadIdx.x; e < 64 * 32; e += 256) {
      int r = e >> 5, c = e & 31;
      tileB[e] = W[(size_t)(n0 + r) * Ktot + k0 + c];
    }
    __syncthreads();
    const __bf16* tb = &tileB[0];
#endif

    // ---- A fragment (f32 -> bf16 on the fly), segment select on K-concat ----
    const float* Ap; int col;
    if (k0 < K0) { Ap = A0 + (size_t)am * lda0; col = k0; }
    else         { Ap = A1 + (size_t)am * lda1; col = k0 - K0; }
    __builtin_prefetch(Ap + col + 32, 0, 0);    // next A chunk
    f32x8 x0 = *reinterpret_cast<const f32x8*>(Ap + col + akb);
    f32x8 x1 = *reinterpret_cast<const f32x8*>(Ap + col + akb + 8);
    v16bf a;
#pragma unroll
    for (int j = 0; j < 8; ++j) { a[j] = (__bf16)x0[j]; a[8 + j] = (__bf16)x1[j]; }

    // ---- load all 4 B fragments up front, then 4 back-to-back WMMAs ----
    const v16bf b0v = *reinterpret_cast<const v16bf*>(&tb[( 0 + (lane & 15)) * 32 + bk16]);
    const v16bf b1v = *reinterpret_cast<const v16bf*>(&tb[(16 + (lane & 15)) * 32 + bk16]);
    const v16bf b2v = *reinterpret_cast<const v16bf*>(&tb[(32 + (lane & 15)) * 32 + bk16]);
    const v16bf b3v = *reinterpret_cast<const v16bf*>(&tb[(48 + (lane & 15)) * 32 + bk16]);
    acc[0] = __builtin_amdgcn_wmma_f32_16x16x32_bf16(false, a, false, b0v, (short)0, acc[0], false, false);
    acc[1] = __builtin_amdgcn_wmma_f32_16x16x32_bf16(false, a, false, b1v, (short)0, acc[1], false, false);
    acc[2] = __builtin_amdgcn_wmma_f32_16x16x32_bf16(false, a, false, b2v, (short)0, acc[2], false, false);
    acc[3] = __builtin_amdgcn_wmma_f32_16x16x32_bf16(false, a, false, b3v, (short)0, acc[3], false, false);

#if USE_TDM
    // next buffer must be resident before the next iteration reads it
    __builtin_amdgcn_s_wait_tensorcnt(0);
    __syncthreads();
#else
    __syncthreads();
#endif
  }

  // ---- store C/D (+bias): lane half selects M offset 0/8 (ISA C layout) ----
#pragma unroll
  for (int tt = 0; tt < 4; ++tt) {
    const int n = n0 + 16 * tt + (lane & 15);
    const float bn = bias[n];
#pragma unroll
    for (int r = 0; r < 8; ++r) {
      const int m = m0 + (lane >> 4) * 8 + r;
      G[(size_t)m * L4_ + n] = acc[tt][r] + bn;
    }
  }
}

// ---------------------------------------------------------------------------
// LSTM pointwise: gates -> (h, c); optional bf16 copy of h (FC input history)
// ---------------------------------------------------------------------------
__device__ __forceinline__ float sigmoidf_(float x) { return 1.0f / (1.0f + __expf(-x)); }

__global__ void k_lstm_pointwise(const float* __restrict__ G,
                                 float* __restrict__ h, float* __restrict__ c,
                                 __bf16* __restrict__ ybf)
{
  int idx = blockIdx.x * blockDim.x + threadIdx.x;
  if (idx >= B_ * H_) return;
  int b = idx >> 10, j = idx & (H_ - 1);
  const float* g = G + (size_t)b * L4_;
  float vi = g[j], vf = g[j + H_], vg = g[j + 2 * H_], vo = g[j + 3 * H_];
  float cn = sigmoidf_(vf) * c[idx] + sigmoidf_(vi) * tanhf(vg);
  float hn = sigmoidf_(vo) * tanhf(cn);
  c[idx] = cn;
  h[idx] = hn;
  if (ybf) ybf[idx] = (__bf16)hn;
}

// ---------------------------------------------------------------------------
// FC GEMM: out(b, t, 0:64) = A(M x 1024) @ Wfc^T + bfc, A/Wfc bf16.
// Row m -> (t = t_base + (m>>8), b = m&255). Wave computes 16 x 64.
// ---------------------------------------------------------------------------
__global__ __launch_bounds__(256) void k_fc(
    const __bf16* __restrict__ A, const __bf16* __restrict__ Wfc,
    const float* __restrict__ bfc, float* __restrict__ out,
    int M, int t_base, int t_sel, float* __restrict__ xnext)
{
  const int lane = threadIdx.x & 31;
  const int wave = threadIdx.x >> 5;
  const int m0 = (blockIdx.x * 8 + wave) * 16;
  if (m0 >= M) return;

  const int am   = m0 + (lane & 15);
  const int akb  = (lane >> 4) * 16;
  const int bk16 = (lane >> 4) * 16;

  v8f acc[4] = {};
  const __bf16* Arow = A + (size_t)am * H_;

  for (int k0 = 0; k0 < H_; k0 += 32) {
    __builtin_prefetch(Arow + k0 + 64, 0, 0);   // -> global_prefetch_b8
    v8bf a0 = *reinterpret_cast<const v8bf*>(Arow + k0 + akb);
    v8bf a1 = *reinterpret_cast<const v8bf*>(Arow + k0 + akb + 8);
    v16bf a = __builtin_shufflevector(a0, a1, 0, 1, 2, 3, 4, 5, 6, 7,
                                      8, 9, 10, 11, 12, 13, 14, 15);
    const v16bf b0v = *reinterpret_cast<const v16bf*>(Wfc + (size_t)( 0 + (lane & 15)) * H_ + k0 + bk16);
    const v16bf b1v = *reinterpret_cast<const v16bf*>(Wfc + (size_t)(16 + (lane & 15)) * H_ + k0 + bk16);
    const v16bf b2v = *reinterpret_cast<const v16bf*>(Wfc + (size_t)(32 + (lane & 15)) * H_ + k0 + bk16);
    const v16bf b3v = *reinterpret_cast<const v16bf*>(Wfc + (size_t)(48 + (lane & 15)) * H_ + k0 + bk16);
    acc[0] = __builtin_amdgcn_wmma_f32_16x16x32_bf16(false, a, false, b0v, (short)0, acc[0], false, false);
    acc[1] = __builtin_amdgcn_wmma_f32_16x16x32_bf16(false, a, false, b1v, (short)0, acc[1], false, false);
    acc[2] = __builtin_amdgcn_wmma_f32_16x16x32_bf16(false, a, false, b2v, (short)0, acc[2], false, false);
    acc[3] = __builtin_amdgcn_wmma_f32_16x16x32_bf16(false, a, false, b3v, (short)0, acc[3], false, false);
  }

#pragma unroll
  for (int tt = 0; tt < 4; ++tt) {
    const int n = 16 * tt + (lane & 15);
    const float bn = bfc[n];
#pragma unroll
    for (int r = 0; r < 8; ++r) {
      const int m = m0 + (lane >> 4) * 8 + r;
      const int t = t_base + (m >> 8);
      const int bb = m & (B_ - 1);
      const float v = acc[tt][r] + bn;
      out[(size_t)bb * TO_ * F_ + (size_t)t * F_ + n] = v;
      if (xnext && t == t_sel) xnext[bb * F_ + n] = v;
    }
  }
}

// ---------------------------------------------------------------------------
// Host: workspace layout + launch schedule (graph-capture safe)
// ---------------------------------------------------------------------------
extern "C" void kernel_launch(void* const* d_in, const int* in_sizes, int n_in,
                              void* d_out, int out_size, void* d_ws, size_t ws_size,
                              hipStream_t stream)
{
  (void)in_sizes; (void)n_in; (void)out_size; (void)ws_size;

  const float* input = (const float*)d_in[0];
  const float* W_ih0 = (const float*)d_in[1];
  const float* W_hh0 = (const float*)d_in[2];
  const float* b_ih0 = (const float*)d_in[3];
  const float* b_hh0 = (const float*)d_in[4];
  const float* W_ih1 = (const float*)d_in[5];
  const float* W_hh1 = (const float*)d_in[6];
  const float* b_ih1 = (const float*)d_in[7];
  const float* b_hh1 = (const float*)d_in[8];
  const float* W_fc  = (const float*)d_in[9];
  const float* b_fc  = (const float*)d_in[10];
  float* out = (float*)d_out;

  char* ws = (char*)d_ws;
  size_t off = 0;
  auto alloc = [&](size_t bytes) { size_t o = off; off = (off + bytes + 255) & ~(size_t)255; return o; };

  __bf16* W0cat = (__bf16*)(ws + alloc((size_t)L4_ * (F_ + H_) * 2)); // 4096 x 1088
  __bf16* W1cat = (__bf16*)(ws + alloc((size_t)L4_ * (2 * H_) * 2)); // 4096 x 2048
  __bf16* Wfcbf = (__bf16*)(ws + alloc((size_t)F_ * H_ * 2));        // 64 x 1024
  float*  b0    = (float*)(ws + alloc((size_t)L4_ * 4));
  float*  b1    = (float*)(ws + alloc((size_t)L4_ * 4));
  float*  G     = (float*)(ws + alloc((size_t)B_ * L4_ * 4));
  float*  h0    = (float*)(ws + alloc((size_t)4 * B_ * H_ * 4));     // h0,c0,h1,c1 contiguous
  float*  c0 = h0 + (size_t)B_ * H_;
  float*  h1 = c0 + (size_t)B_ * H_;
  float*  c1 = h1 + (size_t)B_ * H_;
  __bf16* y1bf = (__bf16*)(ws + alloc((size_t)T_ * B_ * H_ * 2));    // 268 MB history
  __bf16* arbf = (__bf16*)(ws + alloc((size_t)B_ * H_ * 2));
  float*  xar  = (float*)(ws + alloc((size_t)B_ * F_ * 4));

  // ---- prep: combined biases, bf16 K-concatenated weights, zeroed state ----
  k_bias_combine<<<16, 256, 0, stream>>>(b_ih0, b_hh0, b_ih1, b_hh1, b0, b1);
  k_wcat_bf16<<<2048, 256, 0, stream>>>(W_ih0, F_, W_hh0, H_, W0cat, L4_);
  k_wcat_bf16<<<2048, 256, 0, stream>>>(W_ih1, H_, W_hh1, H_, W1cat, L4_);
  k_wcat_bf16<<<256, 256, 0, stream>>>(W_fc, H_, nullptr, 0, Wfcbf, F_);
  k_zero_f32<<<1024, 256, 0, stream>>>(h0, (long)4 * B_ * H_);

  const dim3 gate_grid(L4_ / 64, B_ / 128);   // 64 x 2 blocks, 8 waves each

  // ---- main recurrence over T timesteps ----
  for (int t = 0; t < T_; ++t) {
    k_lstm_gates<<<gate_grid, 256, 0, stream>>>(
        input + (size_t)t * F_, T_ * F_, F_, h0, H_, H_, W0cat, b0, G);
    k_lstm_pointwise<<<(B_ * H_) / 256, 256, 0, stream>>>(G, h0, c0, nullptr);
    k_lstm_gates<<<gate_grid, 256, 0, stream>>>(
        h0, H_, H_, h1, H_, H_, W1cat, b1, G);
    k_lstm_pointwise<<<(B_ * H_) / 256, 256, 0, stream>>>(
        G, h1, c1, y1bf + (size_t)t * B_ * H_);
  }

  // ---- one big FC GEMM over the whole history; seed x for AR from t=T-1 ----
  k_fc<<<(B_ * T_) / 128, 256, 0, stream>>>(
      y1bf, Wfcbf, b_fc, out, B_ * T_, 0, T_ - 1, xar);

  // ---- autoregressive future steps ----
  for (int s = 0; s < FUT_; ++s) {
    k_lstm_gates<<<gate_grid, 256, 0, stream>>>(
        xar, F_, F_, h0, H_, H_, W0cat, b0, G);
    k_lstm_pointwise<<<(B_ * H_) / 256, 256, 0, stream>>>(G, h0, c0, nullptr);
    k_lstm_gates<<<gate_grid, 256, 0, stream>>>(
        h0, H_, H_, h1, H_, H_, W1cat, b1, G);
    k_lstm_pointwise<<<(B_ * H_) / 256, 256, 0, stream>>>(G, h1, c1, arbf);
    k_fc<<<B_ / 128, 256, 0, stream>>>(
        arbf, Wfcbf, b_fc, out, B_, T_ + s, T_ + s, xar);
  }
}
